// Pointnet2_7_35828617183509
// MI455X (gfx1250) — matrix operand order repaired and checked by reading, not compiled, not run on hardware
//
#include <hip/hip_runtime.h>
#include <hip/hip_bf16.h>
#include <math.h>

typedef __attribute__((ext_vector_type(16))) _Float16 v16h;
typedef __attribute__((ext_vector_type(8)))  float    v8f;

#define BATCH 2
#define NPTS  16384
#define BN_SCALE 0.9999950000374994f   // 1/sqrt(1+1e-5)

// ============================================================================
// transpose (B,3,N) -> (B,N,3)
// ============================================================================
__global__ void k_transpose(const float* __restrict__ x, float* __restrict__ y) {
  int t = blockIdx.x * blockDim.x + threadIdx.x;
  if (t >= BATCH * NPTS) return;
  int b = t / NPTS, i = t % NPTS;
#pragma unroll
  for (int c = 0; c < 3; ++c)
    y[(size_t)t * 3 + c] = x[((size_t)b * 3 + c) * NPTS + i];
}

// ============================================================================
// Farthest point sampling: one block per batch, serial chain with LDS argmax.
// ============================================================================
__global__ void k_fps(const float* __restrict__ pts, int n, int npoint,
                      float* __restrict__ dist, int* __restrict__ out_idx) {
  int b = blockIdx.x;
  const float* p = pts + (size_t)b * n * 3;
  float* dd = dist + (size_t)b * n;
  int* oi = out_idx + (size_t)b * npoint;
  __shared__ float rmax[256];
  __shared__ int   rarg[256];
  __shared__ int   cur;
  int tid = threadIdx.x;
  for (int i = tid; i < n; i += 256) dd[i] = 1e10f;
  if (tid == 0) cur = 0;
  __syncthreads();
  for (int s = 0; s < npoint; ++s) {
    int c = cur;
    if (tid == 0) oi[s] = c;
    float cx = p[c * 3], cy = p[c * 3 + 1], cz = p[c * 3 + 2];
    float best = -1.0f; int bi = 0;
    for (int i = tid; i < n; i += 256) {
      float dx = p[i * 3] - cx, dy = p[i * 3 + 1] - cy, dz = p[i * 3 + 2] - cz;
      float d = dx * dx + dy * dy + dz * dz;
      float m = fminf(dd[i], d);
      dd[i] = m;
      if (m > best) { best = m; bi = i; }
    }
    rmax[tid] = best; rarg[tid] = bi;
    __syncthreads();
    for (int off = 128; off > 0; off >>= 1) {
      if (tid < off) {
        if (rmax[tid + off] > rmax[tid] ||
            (rmax[tid + off] == rmax[tid] && rarg[tid + off] < rarg[tid])) {
          rmax[tid] = rmax[tid + off]; rarg[tid] = rarg[tid + off];
        }
      }
      __syncthreads();
    }
    if (tid == 0) cur = rarg[0];
    __syncthreads();
  }
}

// gather 3-channel coords by fps index
__global__ void k_gather3(const float* __restrict__ src, const int* __restrict__ idx,
                          int n, int npoint, float* __restrict__ dst) {
  int t = blockIdx.x * blockDim.x + threadIdx.x;
  if (t >= BATCH * npoint) return;
  int b = t / npoint;
  int pi = idx[t];
#pragma unroll
  for (int c = 0; c < 3; ++c)
    dst[(size_t)t * 3 + c] = src[((size_t)b * n + pi) * 3 + c];
}

// ============================================================================
// Ball query: first K point indices (ascending) within radius; pad with first.
// ============================================================================
__global__ void k_query_ball(const float* __restrict__ pts, const float* __restrict__ nxyz,
                             int n, int npoint, float r2, int K, int* __restrict__ idx) {
  int t = blockIdx.x * blockDim.x + threadIdx.x;
  if (t >= BATCH * npoint) return;
  int b = t / npoint;
  const float* q = nxyz + (size_t)t * 3;
  const float* p = pts + (size_t)b * n * 3;
  int* o = idx + (size_t)t * K;
  float qx = q[0], qy = q[1], qz = q[2];
  int cnt = 0, first = 0;
  for (int i = 0; i < n && cnt < K; ++i) {
    float dx = p[i * 3] - qx, dy = p[i * 3 + 1] - qy, dz = p[i * 3 + 2] - qz;
    float d = dx * dx + dy * dy + dz * dz;
    if (d <= r2) { if (cnt == 0) first = i; o[cnt++] = i; }
  }
  for (; cnt < K; ++cnt) o[cnt] = first;
}

// ============================================================================
// Grouping into padded rows: G[row, 0..C)=feat, [C..C+3)=rel xyz, [cin..ld)=0
// ============================================================================
__global__ void k_group(const float* __restrict__ pts, const float* __restrict__ feat,
                        const float* __restrict__ nxyz, const int* __restrict__ idx,
                        int n, int npoint, int K, int C, float* __restrict__ G, int ld) {
  size_t total = (size_t)BATCH * npoint * K * ld;
  size_t e = (size_t)blockIdx.x * blockDim.x + threadIdx.x;
  if (e >= total) return;
  int c = (int)(e % ld);
  size_t row = e / ld;
  int k = (int)(row % K);
  size_t bj = row / K;
  int j = (int)(bj % npoint);
  int b = (int)(bj / npoint);
  float v = 0.0f;
  if (c < C + 3) {
    int pi = idx[((size_t)b * npoint + j) * K + k];
    if (c < C) v = feat[((size_t)b * n + pi) * C + c];
    else       v = pts[((size_t)b * n + pi) * 3 + (c - C)] -
                   nxyz[((size_t)b * npoint + j) * 3 + (c - C)];
  }
  G[row * ld + c] = v;
}

// ============================================================================
// WMMA GEMM: Y[M,Cout] = act((X[M,Cin] . W[Cout,Cin]^T + b) * s), strided/padded.
// Block: 128 threads = 4 waves, 128x16 tile; each wave does two 16x16 subtiles
// sharing the B fragment -> 2 v_wmma per K-step. Uniform fast path stages a
// full 128x32 A tile branch-free (float4 loads -> cvt_pk -> ds_store_b128);
// B staged 8-wide when weight rows are 16B aligned (Cin % 4 == 0).
// ============================================================================
union HalfPack8 { _Float16 h[8]; int4 v4; };

__device__ inline void pack8_store(const float* __restrict__ src, _Float16* dst) {
  const float4* s = (const float4*)src;
  float4 f0 = s[0], f1 = s[1];
  HalfPack8 pk;
  pk.h[0] = (_Float16)f0.x; pk.h[1] = (_Float16)f0.y;
  pk.h[2] = (_Float16)f0.z; pk.h[3] = (_Float16)f0.w;
  pk.h[4] = (_Float16)f1.x; pk.h[5] = (_Float16)f1.y;
  pk.h[6] = (_Float16)f1.z; pk.h[7] = (_Float16)f1.w;
  *(int4*)dst = pk.v4;
}

__launch_bounds__(128)
__global__ void k_gemm_wmma(const float* __restrict__ X, int ldX,
                            const float* __restrict__ W,
                            const float* __restrict__ bias, float* __restrict__ Y,
                            int ldY, int M, int Cin, int Cout, int act) {
  __shared__ v16h lds_a[8][32];   // 8 subtiles x (lane -> 16 halves)
  __shared__ v16h lds_b[32];
  int tid  = threadIdx.x;
  int wave = tid >> 5;
  int lane = tid & 31;
  int mBase = blockIdx.x * 128;
  int n0 = blockIdx.y * 16;
  bool fullM = (mBase + 128 <= M);
  bool bVec  = ((Cin & 3) == 0);

  v8f acc0 = {}, acc1 = {};
  for (int k0 = 0; k0 < ldX; k0 += 32) {
    // ---- A: 128 rows x 32 K as 512 8-wide units (4 per thread) ----
    if (fullM && (k0 + 32 <= ldX)) {
#pragma unroll
      for (int u4 = 0; u4 < 4; ++u4) {
        int u   = tid + u4 * 128;
        int mr  = u >> 2, grp = u & 3;
        int st  = mr >> 4, r = mr & 15;
        int ls  = r + ((grp & 1) << 4);
        int hb  = (grp & 2) << 2;
        pack8_store(X + (size_t)(mBase + mr) * ldX + k0 + grp * 8,
                    (_Float16*)(&lds_a[st][ls]) + hb);
        if (grp == 0 && k0 + 32 < ldX)
          __builtin_prefetch(X + (size_t)(mBase + mr) * ldX + k0 + 32, 0, 1);
      }
    } else {
      for (int u = tid; u < 512; u += 128) {
        int mr  = u >> 2, grp = u & 3;
        int gm  = mBase + mr;
        int st  = mr >> 4, r = mr & 15;
        int ls  = r + ((grp & 1) << 4);
        int hb  = (grp & 2) << 2;
        _Float16* dst = (_Float16*)(&lds_a[st][ls]) + hb;
        if (gm < M && (k0 + grp * 8 + 8) <= ldX) {
          pack8_store(X + (size_t)gm * ldX + k0 + grp * 8, dst);
        } else {
          HalfPack8 pk; pk.v4 = make_int4(0, 0, 0, 0);
          *(int4*)dst = pk.v4;
        }
      }
    }
    // ---- B: 32 K x 16 N as 64 8-wide units ----
    for (int u = tid; u < 64; u += 128) {
      int n  = u >> 2, kq = u & 3;
      int gn = n0 + n;
      int kb = k0 + kq * 8;
      int ls = n + ((kq & 1) << 4);   // +16 for K in 8..15 / 24..31
      int hb = (kq & 2) << 2;         // halves 0..7 (K < 16) or 8..15 (K >= 16)
      _Float16* dst = (_Float16*)(&lds_b[ls]) + hb;
      if (bVec && gn < Cout && kb + 8 <= Cin) {
        pack8_store(W + (size_t)gn * Cin + kb, dst);
      } else {
        HalfPack8 pk;
#pragma unroll
        for (int j = 0; j < 8; ++j) {
          float v = (gn < Cout && kb + j < Cin) ? W[(size_t)gn * Cin + kb + j] : 0.0f;
          pk.h[j] = (_Float16)v;
        }
        *(int4*)dst = pk.v4;
      }
    }
    __syncthreads();
    v16h a0 = lds_a[wave * 2 + 0][lane];
    v16h a1 = lds_a[wave * 2 + 1][lane];
    v16h bb = lds_b[lane];
    acc0 = __builtin_amdgcn_wmma_f32_16x16x32_f16(false, a0, false, bb,
                                                  (short)0, acc0, false, false);
    acc1 = __builtin_amdgcn_wmma_f32_16x16x32_f16(false, a1, false, bb,
                                                  (short)0, acc1, false, false);
    __syncthreads();
  }
  // ---- epilogue: C/D layout row = r + 8*(lane>>4), col = lane&15 ----
  int col = lane & 15;
  int rb  = (lane >> 4) * 8;
  int gn  = n0 + col;
  if (gn < ldY) {
    float bv = (gn < Cout) ? bias[gn] : 0.0f;
#pragma unroll
    for (int s = 0; s < 2; ++s) {
      int mt = mBase + (wave * 2 + s) * 16 + rb;
#pragma unroll
      for (int r = 0; r < 8; ++r) {
        int gm = mt + r;
        if (gm < M) {
          float v = 0.0f;
          if (gn < Cout) {
            float x = (s ? acc1[r] : acc0[r]) + bv;
            if (act == 0) { x *= BN_SCALE; v = x > 0.0f ? x : 0.0f; }
            else          { v = 1.0f / (1.0f + __expf(-x)); }
          }
          Y[(size_t)gm * ldY + gn] = v;   // pad columns zeroed for next layer
        }
      }
    }
  }
}

// ============================================================================
// Maxpool over K neighbors: F[bj, co+c] = max_k Y[(bj*K+k)*ldY + c]
// ============================================================================
__global__ void k_maxpool(const float* __restrict__ Y, int ldY, int npoint, int K,
                          int C, float* __restrict__ F, int Ctot, int co) {
  size_t total = (size_t)BATCH * npoint * C;
  size_t e = (size_t)blockIdx.x * blockDim.x + threadIdx.x;
  if (e >= total) return;
  int c = (int)(e % C);
  size_t bj = e / C;
  float m = -1e30f;
  for (int k = 0; k < K; ++k)
    m = fmaxf(m, Y[((size_t)bj * K + k) * ldY + c]);
  F[(size_t)bj * Ctot + co + c] = m;
}

// copy src[B,n,C] (packed) into dst[B,n,ldtot] at channel offset co
__global__ void k_copy_feat(const float* __restrict__ src, int n, int C,
                            float* __restrict__ dst, int ldtot, int co) {
  size_t total = (size_t)BATCH * n * C;
  size_t e = (size_t)blockIdx.x * blockDim.x + threadIdx.x;
  if (e >= total) return;
  int c = (int)(e % C);
  size_t r = e / C;
  dst[r * ldtot + co + c] = src[e];
}

// ============================================================================
// 3-NN inverse-distance interpolation; writes [co..co+C2) and zero pad to ldtot
// ============================================================================
__global__ void k_three_interp(const float* __restrict__ xyz1, const float* __restrict__ xyz2,
                               const float* __restrict__ f2, int ld2, int n1, int n2,
                               int C2, float* __restrict__ dst, int ldtot, int co) {
  int t = blockIdx.x * blockDim.x + threadIdx.x;
  if (t >= BATCH * n1) return;
  int b = t / n1;
  const float* q  = xyz1 + (size_t)t * 3;
  const float* p2 = xyz2 + (size_t)b * n2 * 3;
  float qx = q[0], qy = q[1], qz = q[2];
  float d0 = 1e30f, d1 = 1e30f, d2 = 1e30f;
  int   i0 = 0, i1 = 0, i2 = 0;
  for (int j = 0; j < n2; ++j) {
    float dx = p2[j * 3] - qx, dy = p2[j * 3 + 1] - qy, dz = p2[j * 3 + 2] - qz;
    float d = dx * dx + dy * dy + dz * dz;
    if (d < d0)      { d2 = d1; i2 = i1; d1 = d0; i1 = i0; d0 = d; i0 = j; }
    else if (d < d1) { d2 = d1; i2 = i1; d1 = d;  i1 = j; }
    else if (d < d2) { d2 = d;  i2 = j; }
  }
  float w0 = 1.0f / (d0 + 1e-8f), w1 = 1.0f / (d1 + 1e-8f), w2 = 1.0f / (d2 + 1e-8f);
  float ws = w0 + w1 + w2;
  w0 /= ws; w1 /= ws; w2 /= ws;
  const float* fb = f2 + (size_t)b * n2 * ld2;
  float* o = dst + (size_t)t * ldtot + co;
  int span = ldtot - co;
  for (int c = 0; c < span; ++c) {
    float v = 0.0f;
    if (c < C2)
      v = w0 * fb[(size_t)i0 * ld2 + c] + w1 * fb[(size_t)i1 * ld2 + c] +
          w2 * fb[(size_t)i2 * ld2 + c];
    o[c] = v;
  }
}

// ============================================================================
// Host orchestration
// ============================================================================
extern "C" void kernel_launch(void* const* d_in, const int* in_sizes, int n_in,
                              void* d_out, int out_size, void* d_ws, size_t ws_size,
                              hipStream_t stream) {
  (void)in_sizes; (void)n_in; (void)out_size; (void)ws_size;

  static const float SA_R[7][2]      = {{0.0125f,0.025f},{0.025f,0.05f},{0.05f,0.1f},
                                        {0.1f,0.2f},{0.2f,0.4f},{0.4f,0.8f},{0.8f,1.0f}};
  static const int   SA_K[7][2]      = {{16,32},{16,32},{16,32},{16,32},{16,32},{16,32},{16,16}};
  static const int   SA_MLP[7][2][3] = {
      {{4,4,8},{8,8,16}}, {{8,8,16},{16,16,32}}, {{16,16,32},{32,32,64}},
      {{64,64,128},{64,96,128}}, {{128,196,256},{128,192,256}},
      {{256,256,512},{256,384,512}}, {{512,512,1024},{512,768,1024}}};
  static const int NLEV[8] = {16384, 1024, 512, 512, 256, 64, 16, 8};
  static const int CLEV[8] = {3, 24, 48, 96, 256, 512, 1024, 2048};   // multiples of 8 (l>0)
  static const int FP_NC[7] = {2, 2, 2, 2, 2, 2, 3};
  static const int FP_CO[7][3] = {{512,512,0},{256,256,0},{256,256,0},{256,128,0},
                                  {128,64,0},{64,32,0},{32,32,32}};

  // ---- parameter walk (setup_inputs insertion order: xyz, sa, fp, head) ----
  int ip = 0;
  const float* xyz = (const float*)d_in[ip++];
  const float *saw[7][2][3], *sab[7][2][3];
  for (int l = 0; l < 7; ++l)
    for (int br = 0; br < 2; ++br)
      for (int cv = 0; cv < 3; ++cv) {
        saw[l][br][cv] = (const float*)d_in[ip++];
        sab[l][br][cv] = (const float*)d_in[ip++];
      }
  const float *fpw[7][3], *fpb[7][3];
  for (int i = 0; i < 7; ++i)
    for (int cv = 0; cv < FP_NC[i]; ++cv) {
      fpw[i][cv] = (const float*)d_in[ip++];
      fpb[i][cv] = (const float*)d_in[ip++];
    }
  const float* hw1 = (const float*)d_in[ip++];
  const float* hb1 = (const float*)d_in[ip++];
  const float* hw2 = (const float*)d_in[ip++];
  const float* hb2 = (const float*)d_in[ip++];

  // ---- workspace bump allocator (256B aligned blocks) ----
  char* wsb = (char*)d_ws;
  size_t off = 0;
  auto alloc = [&](size_t bytes) -> void* {
    void* p = wsb + off;
    off = (off + bytes + 255) & ~(size_t)255;
    return p;
  };
  auto pad8 = [](int c) { return (c + 7) & ~7; };
  auto ceil_div = [](size_t a, size_t b) { return (int)((a + b - 1) / b); };

  float* xs[8]; float* fs[8];
  for (int l = 0; l < 8; ++l) xs[l] = (float*)alloc((size_t)BATCH * NLEV[l] * 3 * 4);
  fs[0] = xs[0];
  for (int l = 1; l < 8; ++l) fs[l] = (float*)alloc((size_t)BATCH * NLEV[l] * CLEV[l] * 4);
  float* fpsd    = (float*)alloc((size_t)BATCH * NPTS * 4);
  int*   fps_idx = (int*)  alloc((size_t)BATCH * 1024 * 4);
  int*   ball    = (int*)  alloc((size_t)BATCH * 1024 * 32 * 4);
  float* G       = (float*)alloc((size_t)2 * 1024 * 1024 * 4);
  float* ping    = (float*)alloc((size_t)2560 * 1024 * 4);
  float* pong    = (float*)alloc((size_t)2560 * 1024 * 4);

  k_transpose<<<ceil_div((size_t)BATCH * NPTS, 256), 256, 0, stream>>>(xyz, xs[0]);

  // ---- SA levels ----
  for (int l = 0; l < 7; ++l) {
    int nprev = NLEV[l], np = NLEV[l + 1];
    int Cprev = CLEV[l], Ctot = CLEV[l + 1];
    k_fps<<<BATCH, 256, 0, stream>>>(xs[l], nprev, np, fpsd, fps_idx);
    k_gather3<<<ceil_div((size_t)BATCH * np, 256), 256, 0, stream>>>(
        xs[l], fps_idx, nprev, np, xs[l + 1]);
    int co = 0;
    for (int br = 0; br < 2; ++br) {
      float r = SA_R[l][br];
      int K = SA_K[l][br];
      k_query_ball<<<ceil_div((size_t)BATCH * np, 256), 256, 0, stream>>>(
          xs[l], xs[l + 1], nprev, np, r * r, K, ball);
      int cin = Cprev + 3, ldG = pad8(cin);
      size_t rows = (size_t)BATCH * np * K;
      k_group<<<ceil_div(rows * ldG, 256), 256, 0, stream>>>(
          xs[l], fs[l], xs[l + 1], ball, nprev, np, K, Cprev, G, ldG);
      const float* xin = G;
      int ci = cin, ldi = ldG;
      for (int cv = 0; cv < 3; ++cv) {
        int coc = SA_MLP[l][br][cv], ldo = pad8(coc);
        float* xout = (cv & 1) ? pong : ping;
        dim3 g(ceil_div(rows, 128), ceil_div((size_t)coc, 16));
        k_gemm_wmma<<<g, 128, 0, stream>>>(xin, ldi, saw[l][br][cv], sab[l][br][cv],
                                           xout, ldo, (int)rows, ci, coc, 0);
        xin = xout; ci = coc; ldi = ldo;
      }
      k_maxpool<<<ceil_div((size_t)BATCH * np * ci, 256), 256, 0, stream>>>(
          xin, ldi, np, K, ci, fs[l + 1], Ctot, co);
      co += ci;
    }
  }

  // ---- FP levels ----
  float* fcur = fs[7];
  int Ccur = CLEV[7], ldcur = CLEV[7];
  for (int i = 0; i < 7; ++i) {
    int lvl = 6 - i;
    int n1 = NLEV[lvl], n2 = NLEV[lvl + 1];
    int C1 = (lvl > 0) ? CLEV[lvl] : 0;
    int cin = C1 + Ccur, ldcat = pad8(cin);
    float* X = G;
    if (C1 > 0)
      k_copy_feat<<<ceil_div((size_t)BATCH * n1 * C1, 256), 256, 0, stream>>>(
          fs[lvl], n1, C1, X, ldcat, 0);
    k_three_interp<<<ceil_div((size_t)BATCH * n1, 256), 256, 0, stream>>>(
        xs[lvl], xs[lvl + 1], fcur, ldcur, n1, n2, Ccur, X, ldcat, C1);
    const float* xin = X;
    int ci = cin, ldi = ldcat;
    size_t rows = (size_t)BATCH * n1;
    for (int cv = 0; cv < FP_NC[i]; ++cv) {
      int coc = FP_CO[i][cv], ldo = pad8(coc);
      float* xout = (cv & 1) ? pong : ping;
      dim3 g(ceil_div(rows, 128), ceil_div((size_t)coc, 16));
      k_gemm_wmma<<<g, 128, 0, stream>>>(xin, ldi, fpw[i][cv], fpb[i][cv],
                                         xout, ldo, (int)rows, ci, coc, 0);
      xin = xout; ci = coc; ldi = ldo;
    }
    fcur = (float*)xin; Ccur = ci; ldcur = ldi;
  }

  // ---- head: 32 -> 16 (relu*scale) -> 3 (sigmoid, straight into d_out) ----
  size_t rows = (size_t)BATCH * NPTS;
  {
    dim3 g(ceil_div(rows, 128), 1);
    k_gemm_wmma<<<g, 128, 0, stream>>>(fcur, ldcur, hw1, hb1, pong, 16,
                                       (int)rows, Ccur, 16, 0);
  }
  {
    dim3 g(ceil_div(rows, 128), 1);
    k_gemm_wmma<<<g, 128, 0, stream>>>(pong, 16, hw2, hb2, (float*)d_out, 3,
                                       (int)rows, 16, 3, 1);
  }
}
